// BaselineSNN_4913442587047
// MI455X (gfx1250) — matrix operand order repaired and checked by reading, not compiled, not run on hardware
//
#include <hip/hip_runtime.h>

#define T_STEPS 100
#define BATCH   1024
#define DIN     784
#define HID     800
#define DOUT    10
#define NPAD    896   // HID padded up to grid tile (7 * 128)
#define SPAD    896   // padded row stride for spikes / membrane state

typedef __attribute__((ext_vector_type(16))) __bf16 v16bf;
typedef __attribute__((ext_vector_type(8)))  float  v8f;

union BFrag { v16bf v; uint4 q[2]; };

__device__ __forceinline__ __bf16 f2bf(float f) {
    union { float f; unsigned u; } x; x.f = f;
    unsigned r = (x.u + 0x7FFFu + ((x.u >> 16) & 1u)) >> 16;
    unsigned short h = (unsigned short)r;
    return __builtin_bit_cast(__bf16, h);
}

__global__ void zero_f32_kernel(float* p, int n) {
    int i = blockIdx.x * blockDim.x + threadIdx.x;
    if (i < n) p[i] = 0.f;
}

// dst[rd x cd] (bf16, row-major) = src[rs x cs] zero-padded
__global__ void cvt_pad_kernel(const float* __restrict__ src, __bf16* __restrict__ dst,
                               int rs, int cs, int rd, int cd) {
    int i = blockIdx.x * blockDim.x + threadIdx.x;
    if (i >= rd * cd) return;
    int r = i / cd, c = i % cd;
    float v = (r < rs && c < cs) ? src[(size_t)r * cs + c] : 0.f;
    dst[i] = f2bf(v);
}

// ---------------------------------------------------------------------------
// GEMM (C = A @ W^T + b) fused with LIF update, layers 1 & 2.
// Block: 256 threads = 8 waves, waves tiled 2(M) x 4(N), 32x32 per wave.
// Grid: (1024/64, 896/128) = (16, 7). K = 800 (layer1: x zero-padded 784->800).
// ---------------------------------------------------------------------------
template<bool LAYER1>
__global__ __launch_bounds__(256)
void gemm_lif_wide(const void* __restrict__ Aptr,
                   const __bf16* __restrict__ W,      // [NPAD x HID] bf16
                   const float* __restrict__ bias,    // [HID]
                   const float* __restrict__ beta_p,  // scalar
                   float* __restrict__ mstate,        // [BATCH x SPAD]
                   float* __restrict__ spk_f32,       // [BATCH x HID]  (d_out slice)
                   __bf16* __restrict__ spk_b16) {    // [BATCH x SPAD]
    const int lane  = threadIdx.x & 31;
    const int wid   = threadIdx.x >> 5;
    const int kgrp  = lane >> 4;     // 0/1: which K half this lane holds
    const int l16   = lane & 15;
    const int waveM = wid & 1;
    const int waveN = wid >> 1;
    const int m0 = blockIdx.x * 64 + waveM * 32;
    const int n0 = blockIdx.y * 128 + waveN * 32;
    const int khalf = kgrp * 8;

    const v8f z = {0.f,0.f,0.f,0.f,0.f,0.f,0.f,0.f};
    v8f c00 = z, c01 = z, c10 = z, c11 = z;

    const int arow0 = m0 + l16;
    const int arow1 = m0 + 16 + l16;
    const int bn0   = n0 + l16;
    const int bn1   = n0 + 16 + l16;

    for (int kk = 0; kk < HID; kk += 32) {
        BFrag a0, a1, b0, b1;
        if (LAYER1) {
            const float* A = (const float*)Aptr;   // [BATCH x DIN] fp32
            #pragma unroll
            for (int g = 0; g < 2; ++g) {
                const int kb = kk + g * 16 + khalf;        // 8-aligned group base
                const bool ok = (kb < DIN);                 // whole group valid or not
                float4 f00 = make_float4(0.f,0.f,0.f,0.f), f01 = f00;
                float4 f10 = f00, f11 = f00;
                if (ok) {
                    const float4* p0 = (const float4*)(A + (size_t)arow0 * DIN + kb);
                    const float4* p1 = (const float4*)(A + (size_t)arow1 * DIN + kb);
                    f00 = p0[0]; f01 = p0[1];
                    f10 = p1[0]; f11 = p1[1];
                }
                a0.v[g*8+0] = f2bf(f00.x); a0.v[g*8+1] = f2bf(f00.y);
                a0.v[g*8+2] = f2bf(f00.z); a0.v[g*8+3] = f2bf(f00.w);
                a0.v[g*8+4] = f2bf(f01.x); a0.v[g*8+5] = f2bf(f01.y);
                a0.v[g*8+6] = f2bf(f01.z); a0.v[g*8+7] = f2bf(f01.w);
                a1.v[g*8+0] = f2bf(f10.x); a1.v[g*8+1] = f2bf(f10.y);
                a1.v[g*8+2] = f2bf(f10.z); a1.v[g*8+3] = f2bf(f10.w);
                a1.v[g*8+4] = f2bf(f11.x); a1.v[g*8+5] = f2bf(f11.y);
                a1.v[g*8+6] = f2bf(f11.z); a1.v[g*8+7] = f2bf(f11.w);
            }
        } else {
            const __bf16* A = (const __bf16*)Aptr; // [BATCH x SPAD] bf16 spikes
            const int kb0 = kk + khalf, kb1 = kk + 16 + khalf;
            a0.q[0] = *(const uint4*)(A + (size_t)arow0 * SPAD + kb0);
            a0.q[1] = *(const uint4*)(A + (size_t)arow0 * SPAD + kb1);
            a1.q[0] = *(const uint4*)(A + (size_t)arow1 * SPAD + kb0);
            a1.q[1] = *(const uint4*)(A + (size_t)arow1 * SPAD + kb1);
        }
        const int kb = kk + khalf;
        b0.q[0] = *(const uint4*)(W + (size_t)bn0 * HID + kb);
        b0.q[1] = *(const uint4*)(W + (size_t)bn0 * HID + kb + 16);
        b1.q[0] = *(const uint4*)(W + (size_t)bn1 * HID + kb);
        b1.q[1] = *(const uint4*)(W + (size_t)bn1 * HID + kb + 16);

        c00 = __builtin_amdgcn_wmma_f32_16x16x32_bf16(false, a0.v, false, b0.v, (short)0, c00, false, false);
        c01 = __builtin_amdgcn_wmma_f32_16x16x32_bf16(false, a0.v, false, b1.v, (short)0, c01, false, false);
        c10 = __builtin_amdgcn_wmma_f32_16x16x32_bf16(false, a1.v, false, b0.v, (short)0, c10, false, false);
        c11 = __builtin_amdgcn_wmma_f32_16x16x32_bf16(false, a1.v, false, b1.v, (short)0, c11, false, false);
    }

    float beta = beta_p[0];
    beta = fminf(fmaxf(beta, 0.f), 1.f);

    auto epi = [&](const v8f& acc, int r, int cb) {
        const int nn = n0 + cb * 16 + l16;
        const float bi = (nn < HID) ? bias[nn] : 0.f;
        #pragma unroll
        for (int v = 0; v < 8; ++v) {
            const int mm = m0 + r * 16 + v + 8 * kgrp;
            const size_t si = (size_t)mm * SPAD + nn;
            const float cur = acc[v] + bi;
            const float mo  = mstate[si];
            const float rst = (mo > 1.f) ? 1.f : 0.f;
            const float mn  = beta * mo + cur - rst;
            const float sp  = (mn > 1.f) ? 1.f : 0.f;
            mstate[si]  = mn;
            spk_b16[si] = f2bf(sp);
            if (nn < HID) spk_f32[(size_t)mm * HID + nn] = sp;
        }
    };
    epi(c00, 0, 0); epi(c01, 0, 1); epi(c10, 1, 0); epi(c11, 1, 1);
}

// ---------------------------------------------------------------------------
// Layer 3: N padded 10->16, one 16x16 tile per wave, 8 waves -> 128 rows/block.
// Grid: 8 blocks. Also accumulates the time-sum of s3 via atomics.
// ---------------------------------------------------------------------------
__global__ __launch_bounds__(256)
void gemm_lif_out(const __bf16* __restrict__ A,      // [BATCH x SPAD] spikes (s2)
                  const __bf16* __restrict__ W,      // [16 x HID]
                  const float* __restrict__ bias,    // [DOUT]
                  const float* __restrict__ beta_p,
                  float* __restrict__ mstate,        // [BATCH x 16]
                  float* __restrict__ s3r,           // t slice, stride DOUT
                  float* __restrict__ s3sum) {       // [BATCH x DOUT]
    const int lane = threadIdx.x & 31;
    const int wid  = threadIdx.x >> 5;
    const int kgrp = lane >> 4;
    const int l16  = lane & 15;
    const int m0   = blockIdx.x * 128 + wid * 16;
    const int khalf = kgrp * 8;
    const int arow = m0 + l16;

    v8f c = {0.f,0.f,0.f,0.f,0.f,0.f,0.f,0.f};
    for (int kk = 0; kk < HID; kk += 32) {
        BFrag a, b;
        a.q[0] = *(const uint4*)(A + (size_t)arow * SPAD + kk + khalf);
        a.q[1] = *(const uint4*)(A + (size_t)arow * SPAD + kk + 16 + khalf);
        b.q[0] = *(const uint4*)(W + (size_t)l16 * HID + kk + khalf);
        b.q[1] = *(const uint4*)(W + (size_t)l16 * HID + kk + 16 + khalf);
        c = __builtin_amdgcn_wmma_f32_16x16x32_bf16(false, a.v, false, b.v, (short)0, c, false, false);
    }

    float beta = beta_p[0];
    beta = fminf(fmaxf(beta, 0.f), 1.f);
    const int nn = l16;
    const float bi = (nn < DOUT) ? bias[nn] : 0.f;
    #pragma unroll
    for (int v = 0; v < 8; ++v) {
        const int mm = m0 + v + 8 * kgrp;
        const size_t si = (size_t)mm * 16 + nn;
        const float cur = c[v] + bi;
        const float mo  = mstate[si];
        const float rst = (mo > 1.f) ? 1.f : 0.f;
        const float mn  = beta * mo + cur - rst;
        const float sp  = (mn > 1.f) ? 1.f : 0.f;
        mstate[si] = mn;
        if (nn < DOUT) {
            s3r[(size_t)mm * DOUT + nn] = sp;
            atomicAdd(&s3sum[(size_t)mm * DOUT + nn], sp);
        }
    }
}

extern "C" void kernel_launch(void* const* d_in, const int* in_sizes, int n_in,
                              void* d_out, int out_size, void* d_ws, size_t ws_size,
                              hipStream_t stream) {
    (void)in_sizes; (void)n_in; (void)out_size; (void)ws_size;
    const float* x     = (const float*)d_in[0];
    const float* W1    = (const float*)d_in[1];
    const float* b1    = (const float*)d_in[2];
    const float* W2    = (const float*)d_in[3];
    const float* b2    = (const float*)d_in[4];
    const float* W3    = (const float*)d_in[5];
    const float* b3    = (const float*)d_in[6];
    const float* beta1 = (const float*)d_in[7];
    const float* beta2 = (const float*)d_in[8];
    const float* beta3 = (const float*)d_in[9];

    float* out   = (float*)d_out;
    float* s3sum = out;                                   // [BATCH*DOUT]
    float* s1r   = out + (size_t)BATCH * DOUT;            // [T,BATCH,HID]
    float* s2r   = s1r + (size_t)T_STEPS * BATCH * HID;   // [T,BATCH,HID]
    float* s3r   = s2r + (size_t)T_STEPS * BATCH * HID;   // [T,BATCH,DOUT]

    char* ws = (char*)d_ws;
    size_t off = 0;
    auto take = [&](size_t bytes) -> char* {
        char* p = ws + off;
        off += (bytes + 255) & ~(size_t)255;
        return p;
    };
    __bf16* w1b = (__bf16*)take((size_t)NPAD * HID * 2);
    __bf16* w2b = (__bf16*)take((size_t)NPAD * HID * 2);
    __bf16* w3b = (__bf16*)take((size_t)16 * HID * 2);
    __bf16* s1b = (__bf16*)take((size_t)BATCH * SPAD * 2);
    __bf16* s2b = (__bf16*)take((size_t)BATCH * SPAD * 2);
    float*  m1  = (float*)take((size_t)BATCH * SPAD * 4);
    float*  m2  = (float*)take((size_t)BATCH * SPAD * 4);
    float*  m3  = (float*)take((size_t)BATCH * 16 * 4);

    int n;
    n = NPAD * HID;
    cvt_pad_kernel<<<(n + 255) / 256, 256, 0, stream>>>(W1, w1b, HID, DIN, NPAD, HID);
    cvt_pad_kernel<<<(n + 255) / 256, 256, 0, stream>>>(W2, w2b, HID, HID, NPAD, HID);
    n = 16 * HID;
    cvt_pad_kernel<<<(n + 255) / 256, 256, 0, stream>>>(W3, w3b, DOUT, HID, 16, HID);
    n = BATCH * SPAD;
    zero_f32_kernel<<<(n + 255) / 256, 256, 0, stream>>>(m1, n);
    zero_f32_kernel<<<(n + 255) / 256, 256, 0, stream>>>(m2, n);
    n = BATCH * 16;
    zero_f32_kernel<<<(n + 255) / 256, 256, 0, stream>>>(m3, n);
    n = BATCH * DOUT;
    zero_f32_kernel<<<(n + 255) / 256, 256, 0, stream>>>(s3sum, n);

    const dim3 gw(16, 7), bw(256);
    const dim3 g3(8), bb(256);
    for (int t = 0; t < T_STEPS; ++t) {
        const float* xt = x + (size_t)t * BATCH * DIN;
        gemm_lif_wide<true ><<<gw, bw, 0, stream>>>(xt,  w1b, b1, beta1, m1,
                                                    s1r + (size_t)t * BATCH * HID, s1b);
        gemm_lif_wide<false><<<gw, bw, 0, stream>>>(s1b, w2b, b2, beta2, m2,
                                                    s2r + (size_t)t * BATCH * HID, s2b);
        gemm_lif_out<<<g3, bb, 0, stream>>>(s2b, w3b, b3, beta3, m3,
                                            s3r + (size_t)t * BATCH * DOUT, s3sum);
    }
}
